// ScaledDotProductAttention_26371099197656
// MI455X (gfx1250) — matrix operand order repaired and checked by reading, not compiled, not run on hardware
//
#include <hip/hip_runtime.h>

typedef __attribute__((ext_vector_type(16))) _Float16 v16h;
typedef __attribute__((ext_vector_type(8)))  _Float16 v8h;
typedef __attribute__((ext_vector_type(4)))  _Float16 v4h;
typedef __attribute__((ext_vector_type(8)))  float    v8f;
typedef __attribute__((ext_vector_type(4)))  unsigned int v4u;
typedef __attribute__((ext_vector_type(8)))  int      v8i;
typedef __attribute__((ext_vector_type(4)))  int      v4i;

#define B_ 8
#define S_ 1024
#define H_ 16
#define D_ 64
#define KT_ 32              // keys per k-tile
#define QB_ 128             // queries per block (4 waves x 32 rows)
#define NT_ (S_ / KT_)      // k-tiles
#define NEG_ (-1e10f)
// exp2-domain scale: (1/sqrt(D)) * log2(e)
#define SC2_ 0.18033688011112042f

// Load 16 consecutive f16 from LDS as a v16h fragment (two ds_load_b128).
__device__ __forceinline__ v16h ld16(const _Float16* p) {
  v8h lo = *(const v8h*)p;
  v8h hi = *(const v8h*)(p + 8);
  v16h r;
  #pragma unroll
  for (int i = 0; i < 8; ++i) { r[i] = lo[i]; r[i + 8] = hi[i]; }
  return r;
}

// TDM: DMA a 2D tile (tile_dim0=64 elems x tile_dim1=32 rows, f32,
// row stride = H_*D_ = 1024 elems) from global memory into LDS.
// D# packing per CDNA5 ISA ch.8 (group0 128b, group1 256b); groups 2/3 unused (2D).
// This toolchain's builtin takes 6 args: (v4u, v8i, v4i, v4i, v8i, i32 cpol).
__device__ __forceinline__ void tdm_load_tile(unsigned long long gaddr,
                                              unsigned int lds_off) {
  v4u g0;
  g0[0] = 1u;                                        // count=1 (valid user D#)
  g0[1] = lds_off;                                   // lds_addr (bytes)
  g0[2] = (unsigned int)(gaddr & 0xFFFFFFFFu);       // global_addr[31:0]
  g0[3] = (unsigned int)((gaddr >> 32) & 0x01FFFFFFu) | 0x80000000u; // [56:32]|type=2
  v8i g1;
  g1[0] = 0x00020000;          // workgroup_mask=0, data_size=2 (4 bytes)
  g1[1] = (int)(1024u << 16);  // tensor_dim0[15:0]=1024 in [31:16]
  g1[2] = (int)(4096u << 16);  // tensor_dim0 hi=0 | tensor_dim1[15:0]=4096
  g1[3] = (int)(64u << 16);    // tensor_dim1 hi=0 | tile_dim0=64
  g1[4] = 32;                  // tile_dim1=32, tile_dim2=0
  g1[5] = 1024;                // tensor_dim0_stride = 1024 elems (lo32)
  g1[6] = 0;                   // stride hi16 = 0, tensor_dim1_stride lo16 = 0
  g1[7] = 0;
  v4i z4 = {0, 0, 0, 0};
  v8i z8 = {0, 0, 0, 0, 0, 0, 0, 0};
  __builtin_amdgcn_tensor_load_to_lds(g0, g1, z4, z4, z8, 0);
}

__global__ void __launch_bounds__(128) zero_masks_kernel(int* m, int n) {
  int i = blockIdx.x * blockDim.x + threadIdx.x;
  if (i < n) m[i] = 0;
}

__global__ void __launch_bounds__(128) scatter_masks_kernel(
    const int* __restrict__ qmb, const int* __restrict__ qms,
    const int* __restrict__ kmb, const int* __restrict__ kms,
    int* __restrict__ qmask, int* __restrict__ kmask, int n) {
  int i = blockIdx.x * blockDim.x + threadIdx.x;
  if (i < n) {
    qmask[qmb[i] * S_ + qms[i]] = 1;  // masked query row (all k -> NEG)
    kmask[kmb[i] * S_ + kms[i]] = 1;  // masked key column (all q -> NEG)
  }
}

__global__ void __launch_bounds__(128) flash_attn_kernel(
    const float* __restrict__ Q, const float* __restrict__ K,
    const float* __restrict__ V, const int* __restrict__ qmask,
    const int* __restrict__ kmask, float* __restrict__ out) {
  // TDM f32 staging (linear row-major tile)
  __shared__ float Kf[KT_ * D_];          // 8 KB
  __shared__ float Vf[KT_ * D_];          // 8 KB
  // f16 compute tiles; padded strides so 16 consecutive rows hit distinct banks
  __shared__ _Float16 Kt[KT_][72];        // K tile [key][d]
  __shared__ _Float16 Vt[D_][40];         // V tile transposed [d][key]
  __shared__ _Float16 Pb[4][2][16][40];   // per-wave, per-qtile P staging

  const int tid  = threadIdx.x;
  const int wave = tid >> 5;
  const int lane = tid & 31;
  const int l16  = lane & 15;
  const int hl   = lane >> 4;             // half-wave id (WMMA layout split)

  const int bh = blockIdx.y;
  const int b  = bh >> 4;
  const int h  = bh & 15;
  const int qwave = blockIdx.x * QB_ + wave * 32;   // 32 q rows per wave

  // ---- Q A-fragments: 2 q-tiles x (16x64 as two 16x32 halves) ----
  // A k order used here (B matches): lane = row M; element i -> K = i+16*hl
  v16h aQ[2][2];
  #pragma unroll
  for (int qt = 0; qt < 2; ++qt) {
    const int qrow = qwave + qt * 16 + l16;
    const float* qptr = Q + (((long)b * S_ + qrow) * H_ + h) * D_;
    #pragma unroll
    for (int kq = 0; kq < 2; ++kq)
      #pragma unroll
      for (int i = 0; i < 16; ++i) {
        int kk = kq * 32 + hl * 16 + i;
        aQ[qt][kq][i] = (_Float16)qptr[kk];
      }
  }

  // row selector: +inf (keep) or NEG (masked query row)
  const float INF_ = __builtin_inff();
  float rowsel[2][8];
  #pragma unroll
  for (int qt = 0; qt < 2; ++qt)
    #pragma unroll
    for (int r = 0; r < 8; ++r)
      rowsel[qt][r] =
          qmask[b * S_ + qwave + qt * 16 + r + hl * 8] ? NEG_ : INF_;

  float m_run[2][8], l_run[2][8];
  v8f acc[2][4];
  #pragma unroll
  for (int qt = 0; qt < 2; ++qt) {
    #pragma unroll
    for (int r = 0; r < 8; ++r) { m_run[qt][r] = -3.0e38f; l_run[qt][r] = 0.0f; }
    #pragma unroll
    for (int c = 0; c < 4; ++c) { v8f z = {}; acc[qt][c] = z; }
  }

  const long kvbase = ((long)b * S_) * H_ * D_ + (long)h * D_;
  const unsigned long long kg = (unsigned long long)(const void*)(K + kvbase);
  const unsigned long long vg = (unsigned long long)(const void*)(V + kvbase);
  const unsigned int ldsK = (unsigned int)(size_t)(void*)Kf;
  const unsigned int ldsV = (unsigned int)(size_t)(void*)Vf;
  const unsigned long long tileBytes = (unsigned long long)KT_ * H_ * D_ * 4;

  // prologue: DMA tile 0 (one wave issues; EXEC is ignored by TDM)
  if (wave == 0) {
    tdm_load_tile(kg, ldsK);
    tdm_load_tile(vg, ldsV);
  }

  for (int kt = 0; kt < NT_; ++kt) {
    // ---- wait for DMA'd tile, publish to workgroup ----
    if (wave == 0) __builtin_amdgcn_s_wait_tensorcnt(0);
    __syncthreads();

    // ---- LDS f32 -> LDS f16 conversion pass (conflict-free b128 reads) ----
    #pragma unroll
    for (int j = 0; j < 4; ++j) {
      int f = j * 128 + tid;            // float4 index in 32x64 tile
      int key = f >> 4;
      int dq  = (f & 15) * 4;
      float4 kq4 = *(const float4*)&Kf[f * 4];
      float4 vq4 = *(const float4*)&Vf[f * 4];
      v4h kv = {(_Float16)kq4.x, (_Float16)kq4.y,
                (_Float16)kq4.z, (_Float16)kq4.w};
      *(v4h*)&Kt[key][dq] = kv;               // 8B-aligned ds_store_b64
      Vt[dq + 0][key] = (_Float16)vq4.x;      // transposed scatter
      Vt[dq + 1][key] = (_Float16)vq4.y;
      Vt[dq + 2][key] = (_Float16)vq4.z;
      Vt[dq + 3][key] = (_Float16)vq4.w;
    }
    __syncthreads();

    // ---- kick DMA for next tile; overlaps with WMMA work below ----
    if (wave == 0 && kt + 1 < NT_) {
      unsigned long long off = (unsigned long long)(kt + 1) * tileBytes;
      tdm_load_tile(kg + off, ldsK);
      tdm_load_tile(vg + off, ldsV);
    }

    const int kbase = kt * KT_;

    // ---- scores: B-fragments loaded once, reused by both q-tiles ----
    v8f sc[2][2];
    float colsel[2];
    #pragma unroll
    for (int cc = 0; cc < 2; ++cc) {
      const int n = cc * 16 + l16;                // local key (B-matrix column)
      v16h b0 = ld16(&Kt[n][hl * 16]);            // d = 0..31
      v16h b1 = ld16(&Kt[n][32 + hl * 16]);       // d = 32..63
      colsel[cc] = kmask[b * S_ + kbase + cc * 16 + l16] ? NEG_ : INF_;
      #pragma unroll
      for (int qt = 0; qt < 2; ++qt) {
        v8f s = {};
        s = __builtin_amdgcn_wmma_f32_16x16x32_f16(false, aQ[qt][0], false, b0,
                                                   (short)0, s, false, false);
        s = __builtin_amdgcn_wmma_f32_16x16x32_f16(false, aQ[qt][1], false, b1,
                                                   (short)0, s, false, false);
        sc[qt][cc] = s;
      }
    }

    // ---- online softmax in exp2 domain; l kept as per-lane partials ----
    #pragma unroll
    for (int qt = 0; qt < 2; ++qt) {
      float corr[8];
      #pragma unroll
      for (int r = 0; r < 8; ++r) {
        float s0 = fminf(fminf(sc[qt][0][r] * SC2_, colsel[0]), rowsel[qt][r]);
        float s1 = fminf(fminf(sc[qt][1][r] * SC2_, colsel[1]), rowsel[qt][r]);
        float t = fmaxf(s0, s1);                  // row max (must be row-uniform)
        t = fmaxf(t, __shfl_xor(t, 1));
        t = fmaxf(t, __shfl_xor(t, 2));
        t = fmaxf(t, __shfl_xor(t, 4));
        t = fmaxf(t, __shfl_xor(t, 8));
        float mn = fmaxf(m_run[qt][r], t);
        float cr = __builtin_amdgcn_exp2f(m_run[qt][r] - mn);
        m_run[qt][r] = mn;
        float p0 = __builtin_amdgcn_exp2f(s0 - mn);
        float p1 = __builtin_amdgcn_exp2f(s1 - mn);
        sc[qt][0][r] = p0; sc[qt][1][r] = p1;
        l_run[qt][r] = l_run[qt][r] * cr + (p0 + p1);  // per-lane partial sum
        corr[r] = cr;
      }
      #pragma unroll
      for (int c = 0; c < 4; ++c)
        #pragma unroll
        for (int r = 0; r < 8; ++r)
          acc[qt][c][r] *= corr[r];

      // P: C-layout -> LDS (A-layout reload below)
      #pragma unroll
      for (int cc = 0; cc < 2; ++cc)
        #pragma unroll
        for (int r = 0; r < 8; ++r)
          Pb[wave][qt][r + hl * 8][cc * 16 + l16] = (_Float16)sc[qt][cc][r];
    }
    asm volatile("s_wait_dscnt 0" ::: "memory");  // wave-local RAW through LDS

    v16h aP[2];
    #pragma unroll
    for (int qt = 0; qt < 2; ++qt) {
      v8h lo = *(const v8h*)&Pb[wave][qt][l16][hl * 16];
      v8h hi = *(const v8h*)&Pb[wave][qt][l16][hl * 16 + 8];
      #pragma unroll
      for (int i = 0; i < 8; ++i) { aP[qt][i] = lo[i]; aP[qt][i + 8] = hi[i]; }
    }

    // ---- O += P V : V B-fragment loaded once per d-chunk, reused 2x ----
    #pragma unroll
    for (int c = 0; c < 4; ++c) {
      v16h bv = ld16(&Vt[c * 16 + l16][hl * 16]);
      #pragma unroll
      for (int qt = 0; qt < 2; ++qt)
        acc[qt][c] = __builtin_amdgcn_wmma_f32_16x16x32_f16(
            false, aP[qt], false, bv, (short)0, acc[qt][c], false, false);
    }
    __syncthreads();
  }

  // ---- epilogue: finish deferred l reduction, normalize, write out ----
  #pragma unroll
  for (int qt = 0; qt < 2; ++qt)
    #pragma unroll
    for (int r = 0; r < 8; ++r) {
      float l = l_run[qt][r];
      l += __shfl_xor(l, 1);
      l += __shfl_xor(l, 2);
      l += __shfl_xor(l, 4);
      l += __shfl_xor(l, 8);
      float inv = __builtin_amdgcn_rcpf(l);
      int q = qwave + qt * 16 + r + hl * 8;
      #pragma unroll
      for (int c = 0; c < 4; ++c)
        out[(((long)b * S_ + q) * H_ + h) * D_ + c * 16 + l16] =
            acc[qt][c][r] * inv;
    }
}

extern "C" void kernel_launch(void* const* d_in, const int* in_sizes, int n_in,
                              void* d_out, int out_size, void* d_ws, size_t ws_size,
                              hipStream_t stream) {
  const float* Q   = (const float*)d_in[0];
  const float* K   = (const float*)d_in[1];
  const float* V   = (const float*)d_in[2];
  const int*   qmb = (const int*)d_in[3];
  const int*   qms = (const int*)d_in[4];
  const int*   kmb = (const int*)d_in[5];
  const int*   kms = (const int*)d_in[6];
  float* out = (float*)d_out;

  int* qmask = (int*)d_ws;
  int* kmask = qmask + B_ * S_;

  int nz = 2 * B_ * S_;
  zero_masks_kernel<<<(nz + 127) / 128, 128, 0, stream>>>(qmask, nz);

  int nm = in_sizes[3];
  scatter_masks_kernel<<<(nm + 127) / 128, 128, 0, stream>>>(qmb, qms, kmb, kms,
                                                             qmask, kmask, nm);

  flash_attn_kernel<<<dim3(S_ / QB_, B_ * H_), 128, 0, stream>>>(Q, K, V,
                                                                 qmask, kmask, out);
}
